// TaylorLinearNet2_34033320854230
// MI455X (gfx1250) — matrix operand order repaired and checked by reading, not compiled, not run on hardware
//
#include <hip/hip_runtime.h>

// ---------------------------------------------------------------------------
// TaylorLinearNet2: out[b,o] = d0[o] + sum_k cumprod_k( X @ W^T )[b, o*8+k]
//   X: (4096,1024) f32, W = allweight: (8192,1024) f32, d0: (1024) f32
//
// Fast path (needs 48MB of d_ws):
//   1) split kernel: fp32 -> bf16 hi/lo planes (one pass, bandwidth-trivial)
//   2) GEMM kernel: bf16x3 WMMA (f32 accum), staging via
//      GLOBAL_LOAD_ASYNC_TO_LDS_B128 (ASYNCcnt) when the builtin exists.
// Fallback: round-1 fused kernel (in-kernel split).
// ---------------------------------------------------------------------------

typedef __attribute__((ext_vector_type(16))) __bf16 v16bf;
typedef __attribute__((ext_vector_type(8)))  __bf16 v8bf;
typedef __attribute__((ext_vector_type(4)))  __bf16 v4bf;
typedef __attribute__((ext_vector_type(8)))  float  v8f;
typedef __attribute__((ext_vector_type(4)))  int    v4i;

static constexpr int BM   = 128;
static constexpr int BN   = 128;
static constexpr int BK   = 32;
static constexpr int IK   = 1024;
static constexpr int NK   = IK / BK;
static constexpr int OD   = 8192;
static constexpr int OCOL = 1024;
static constexpr int LDK  = BK + 8;               // bf16 row stride: 80B, 16B aligned
static constexpr int PLANE = BM * LDK;            // elems per (buf,plane)
static constexpr int XBYTES = 2 * 2 * PLANE * 2;  // 40960
static constexpr int SMEM_BYTES = 2 * XBYTES;     // 81920
static constexpr int ESTRIDE = 68;                // epilogue strip stride (floats)

#if defined(__has_builtin)
#if __has_builtin(__builtin_amdgcn_global_load_async_to_lds_b128)
#define HAVE_ASYNC_LDS 1
#endif
#if __has_builtin(__builtin_amdgcn_s_wait_asynccnt)
#define HAVE_WAIT_ASYNC 1
#endif
#endif

#define AS1 __attribute__((address_space(1)))
#define AS3 __attribute__((address_space(3)))

__device__ __forceinline__ void async_cp16(__bf16* l, const __bf16* g) {
#if defined(HAVE_ASYNC_LDS)
    // b128 async copy: (global int4* src, lds int4* dst, imm offset, imm cpol)
    __builtin_amdgcn_global_load_async_to_lds_b128(
        (AS1 v4i*)(void*)g, (AS3 v4i*)(void*)l, 0, 0);
#else
    *(v8bf*)l = *(const v8bf*)g;
#endif
}

__device__ __forceinline__ void wait_async0() {
#if defined(HAVE_ASYNC_LDS)
#if defined(HAVE_WAIT_ASYNC)
    __builtin_amdgcn_s_wait_asynccnt(0);
#else
    asm volatile("s_wait_asynccnt 0" ::: "memory");
#endif
#endif
}

__device__ __forceinline__ void cvt_split4(float4 v, v4bf& h, v4bf& l) {
    h[0] = (__bf16)v.x; l[0] = (__bf16)(v.x - (float)h[0]);
    h[1] = (__bf16)v.y; l[1] = (__bf16)(v.y - (float)h[1]);
    h[2] = (__bf16)v.z; l[2] = (__bf16)(v.z - (float)h[2]);
    h[3] = (__bf16)v.w; l[3] = (__bf16)(v.w - (float)h[3]);
}

__device__ __forceinline__ v16bf frag2(const __bf16* p0, const __bf16* p1) {
    union { v16bf v; v8bf h[2]; } u;
    u.h[0] = *(const v8bf*)p0;   // ds_load_b128
    u.h[1] = *(const v8bf*)p1;   // ds_load_b128
    return u.v;
}

__device__ __forceinline__ v8f wmma_bf16(v16bf a, v16bf b, v8f c) {
    return __builtin_amdgcn_wmma_f32_16x16x32_bf16(
        false, a, false, b, (short)0, c, false, false);
}

// ---------------------------------------------------------------------------
// Pre-pass: fp32 -> bf16 hi/lo planes
// ---------------------------------------------------------------------------
__global__ __launch_bounds__(256)
void split_bf16_kernel(const float* __restrict__ src,
                       __bf16* __restrict__ hi, __bf16* __restrict__ lo, int n4)
{
    const int i = blockIdx.x * 256 + threadIdx.x;
    if (i >= n4) return;
    float4 v = ((const float4*)src)[i];
    v4bf h, l;
    cvt_split4(v, h, l);
    ((v4bf*)hi)[i] = h;
    ((v4bf*)lo)[i] = l;
}

// ---------------------------------------------------------------------------
// Fast-path GEMM: pre-split bf16 planes, async global->LDS staging,
// 2x4 wave tiling (4 M-waves x 2 N-waves), bf16x3 WMMA, f32 accum.
// ---------------------------------------------------------------------------
__global__ __launch_bounds__(256)
void taylor_gemm_presplit_kernel(const __bf16* __restrict__ Xh, const __bf16* __restrict__ Xl,
                                 const __bf16* __restrict__ Wh, const __bf16* __restrict__ Wl,
                                 const float* __restrict__ D0, float* __restrict__ Out)
{
    extern __shared__ char smem[];
    __bf16* xs = (__bf16*)smem;              // [2 buf][2 plane][BM][LDK]
    __bf16* ws = (__bf16*)(smem + XBYTES);

    const int tid  = threadIdx.x;
    const int lane = tid & 31;
    const int wid  = tid >> 5;
    const int kh   = lane >> 4;
    const int l16  = lane & 15;
    const int wm   = wid & 3;        // 0..3 : M-wave
    const int wn   = wid >> 2;       // 0..1 : N-wave

    const int bm0 = blockIdx.y * BM;
    const int n0  = blockIdx.x * BN;

    // staging: 128 rows x 4 16B-chunks per plane; 2 chunks/thread/plane
    const int srow = tid >> 1;               // 0..127
    const int cb   = (tid & 1) * 2;          // chunk base 0 or 2

    auto stage = [&](int buf, int k0) {
        __bf16* xh = xs + (buf * 2 + 0) * PLANE;
        __bf16* xl = xh + PLANE;
        __bf16* wh = ws + (buf * 2 + 0) * PLANE;
        __bf16* wl = wh + PLANE;
        const size_t gx = (size_t)(bm0 + srow) * IK + k0;
        const size_t gw = (size_t)(n0  + srow) * IK + k0;
#pragma unroll
        for (int c = 0; c < 2; ++c) {
            const int cc  = cb + c;
            const int off = srow * LDK + cc * 8;
            async_cp16(xh + off, Xh + gx + cc * 8);
            async_cp16(xl + off, Xl + gx + cc * 8);
            async_cp16(wh + off, Wh + gw + cc * 8);
            async_cp16(wl + off, Wl + gw + cc * 8);
        }
    };

    v8f acc[2][4];
    const v8f vz = {0.f, 0.f, 0.f, 0.f, 0.f, 0.f, 0.f, 0.f};
#pragma unroll
    for (int mi = 0; mi < 2; ++mi)
#pragma unroll
        for (int ni = 0; ni < 4; ++ni) acc[mi][ni] = vz;

    stage(0, 0);
    wait_async0();
    __syncthreads();

#pragma unroll 1
    for (int ks = 0; ks < NK; ++ks) {
        const int cur = ks & 1;
        if (ks + 1 < NK) stage((ks + 1) & 1, (ks + 1) * BK);  // overlaps compute

        const __bf16* xh = xs + (cur * 2 + 0) * PLANE;
        const __bf16* xl = xh + PLANE;
        const __bf16* wh = ws + (cur * 2 + 0) * PLANE;
        const __bf16* wl = wh + PLANE;

        v16bf a_hi[2], a_lo[2];
#pragma unroll
        for (int mi = 0; mi < 2; ++mi) {
            const __bf16* arh = xh + (wm * 32 + mi * 16 + l16) * LDK;
            const __bf16* arl = xl + (wm * 32 + mi * 16 + l16) * LDK;
            a_hi[mi] = frag2(arh + kh * 16, arh + kh * 16 + 8);
            a_lo[mi] = frag2(arl + kh * 16, arl + kh * 16 + 8);
        }
#pragma unroll
        for (int ni = 0; ni < 4; ++ni) {
            const __bf16* brh = wh + (wn * 64 + ni * 16 + l16) * LDK + kh * 16;
            const __bf16* brl = wl + (wn * 64 + ni * 16 + l16) * LDK + kh * 16;
            const v16bf b_hi = frag2(brh, brh + 8);
            const v16bf b_lo = frag2(brl, brl + 8);
#pragma unroll
            for (int mi = 0; mi < 2; ++mi) {
                acc[mi][ni] = wmma_bf16(a_hi[mi], b_hi, acc[mi][ni]);
                acc[mi][ni] = wmma_bf16(a_hi[mi], b_lo, acc[mi][ni]);
                acc[mi][ni] = wmma_bf16(a_lo[mi], b_hi, acc[mi][ni]);
            }
        }

        if (ks + 1 < NK) {
            wait_async0();
            __syncthreads();
        }
    }

    // --- epilogue: per-wave 32x64 strip -> cumprod-sum over 8 columns -------
    __syncthreads();  // staging LDS dead; reuse for strips

    float* strip = (float*)smem + wid * 32 * ESTRIDE;
#pragma unroll
    for (int mi = 0; mi < 2; ++mi)
#pragma unroll
        for (int ni = 0; ni < 4; ++ni) {
            const int col  = ni * 16 + l16;
            const int rowb = mi * 16 + 8 * kh;
#pragma unroll
            for (int j = 0; j < 8; ++j)
                strip[(rowb + j) * ESTRIDE + col] = acc[mi][ni][j];
        }
    asm volatile("s_wait_dscnt 0" ::: "memory");  // wave-local LDS RAW fence

    // 256 outputs per wave: 32 rows x 8 o's; 8 per lane
#pragma unroll
    for (int q = 0; q < 8; ++q) {
        const int lin = lane * 8 + q;
        const int r   = lin >> 3;        // 0..31
        const int ol  = lin & 7;         // 0..7
        const float* v = strip + r * ESTRIDE + ol * 8;
        float p = v[0], s = v[0];
#pragma unroll
        for (int i = 1; i < 8; ++i) { p *= v[i]; s += p; }
        const int og = blockIdx.x * 16 + wn * 8 + ol;
        Out[(size_t)(bm0 + wm * 32 + r) * OCOL + og] = D0[og] + s;
    }
}

// ---------------------------------------------------------------------------
// Fallback: round-1 fused kernel (in-kernel fp32->bf16 split), 1x8 tiling.
// ---------------------------------------------------------------------------
__device__ __forceinline__ void cvt_split_store4(__bf16* hp, __bf16* lp, float4 v) {
    v4bf h, l;
    cvt_split4(v, h, l);
    *(v4bf*)hp = h;
    *(v4bf*)lp = l;
}

__global__ __launch_bounds__(256)
void taylor_gemm_fused_kernel(const float* __restrict__ X,
                              const float* __restrict__ W,
                              const float* __restrict__ D0,
                              float* __restrict__ Out)
{
    extern __shared__ char smem[];
    __bf16* xs = (__bf16*)smem;
    __bf16* ws = (__bf16*)(smem + XBYTES);

    const int tid  = threadIdx.x;
    const int lane = tid & 31;
    const int wid  = tid >> 5;
    const int kh   = lane >> 4;
    const int l16  = lane & 15;

    const int bm0 = blockIdx.y * BM;
    const int n0  = blockIdx.x * BN;
    const int o0  = blockIdx.x * (BN / 8);

    const int c4 = tid & 7;
    const int r0 = tid >> 3;

    float4 xr[4], wr[4];
    auto load_tile = [&](int k0) {
#pragma unroll
        for (int p = 0; p < 4; ++p) {
            const int row = r0 + p * 32;
            xr[p] = *(const float4*)(X + (size_t)(bm0 + row) * IK + k0 + c4 * 4);
            wr[p] = *(const float4*)(W + (size_t)(n0  + row) * IK + k0 + c4 * 4);
        }
    };
    auto store_tile = [&](int buf) {
        __bf16* xh = xs + (buf * 2 + 0) * PLANE;
        __bf16* xl = xs + (buf * 2 + 1) * PLANE;
        __bf16* wh = ws + (buf * 2 + 0) * PLANE;
        __bf16* wl = ws + (buf * 2 + 1) * PLANE;
#pragma unroll
        for (int p = 0; p < 4; ++p) {
            const int row = r0 + p * 32;
            const int off = row * LDK + c4 * 4;
            cvt_split_store4(xh + off, xl + off, xr[p]);
            cvt_split_store4(wh + off, wl + off, wr[p]);
        }
    };

    v8f acc[8];
    const v8f vz = {0.f, 0.f, 0.f, 0.f, 0.f, 0.f, 0.f, 0.f};
#pragma unroll
    for (int t = 0; t < 8; ++t) acc[t] = vz;

    load_tile(0);
    store_tile(0);
    __syncthreads();

#pragma unroll 1
    for (int ks = 0; ks < NK; ++ks) {
        const int cur = ks & 1;
        if (ks + 1 < NK) load_tile((ks + 1) * BK);

        const __bf16* xh = xs + (cur * 2 + 0) * PLANE;
        const __bf16* xl = xh + PLANE;
        const __bf16* wh = ws + (cur * 2 + 0) * PLANE;
        const __bf16* wl = wh + PLANE;

        const __bf16* arh = xh + (wid * 16 + l16) * LDK;
        const __bf16* arl = xl + (wid * 16 + l16) * LDK;
        const v16bf a_hi = frag2(arh + kh * 16, arh + kh * 16 + 8);
        const v16bf a_lo = frag2(arl + kh * 16, arl + kh * 16 + 8);

#pragma unroll
        for (int t = 0; t < 8; ++t) {
            const __bf16* brh = wh + (t * 16 + l16) * LDK + kh * 16;
            const __bf16* brl = wl + (t * 16 + l16) * LDK + kh * 16;
            const v16bf b_hi = frag2(brh, brh + 8);
            const v16bf b_lo = frag2(brl, brl + 8);
            acc[t] = wmma_bf16(a_hi, b_hi, acc[t]);
            acc[t] = wmma_bf16(a_hi, b_lo, acc[t]);
            acc[t] = wmma_bf16(a_lo, b_hi, acc[t]);
        }

        if (ks + 1 < NK) {
            store_tile((ks + 1) & 1);
            __syncthreads();
        }
    }

    __syncthreads();
    float* strip = (float*)smem + wid * 16 * 132;
#pragma unroll
    for (int t = 0; t < 8; ++t) {
        const int col = t * 16 + l16;
        const int mb  = 8 * kh;
#pragma unroll
        for (int j = 0; j < 8; ++j)
            strip[(mb + j) * 132 + col] = acc[t][j];
    }
    asm volatile("s_wait_dscnt 0" ::: "memory");

#pragma unroll
    for (int q = 0; q < 8; ++q) {
        const int lin = lane * 8 + q;
        const int r   = lin >> 4;
        const int ol  = lin & 15;
        const float* v = strip + r * 132 + ol * 8;
        float p = v[0], s = v[0];
#pragma unroll
        for (int i = 1; i < 8; ++i) { p *= v[i]; s += p; }
        const int og = o0 + ol;
        Out[(size_t)(bm0 + wid * 16 + r) * OCOL + og] = D0[og] + s;
    }
}

// ---------------------------------------------------------------------------
extern "C" void kernel_launch(void* const* d_in, const int* in_sizes, int n_in,
                              void* d_out, int out_size, void* d_ws, size_t ws_size,
                              hipStream_t stream) {
    (void)in_sizes; (void)n_in; (void)out_size;
    const float* x  = (const float*)d_in[0];
    const float* w  = (const float*)d_in[1];
    const float* d0 = (const float*)d_in[2];
    float* out = (float*)d_out;

    dim3 grid(OD / BN, 4096 / BM, 1);   // 64 x 32 tiles

    const size_t xn = (size_t)4096 * 1024;   // elems
    const size_t wn = (size_t)8192 * 1024;
    const size_t need = (xn + wn) * 2 * 2;   // hi+lo planes, 2B each = 48MB

    if (ws_size >= need) {
        char* wsb = (char*)d_ws;
        __bf16* xh = (__bf16*)(wsb);
        __bf16* xl = (__bf16*)(wsb + xn * 2);
        __bf16* wh = (__bf16*)(wsb + xn * 4);
        __bf16* wl = (__bf16*)(wsb + xn * 4 + wn * 2);
        split_bf16_kernel<<<(int)(xn / 4 / 256), 256, 0, stream>>>(x, xh, xl, (int)(xn / 4));
        split_bf16_kernel<<<(int)(wn / 4 / 256), 256, 0, stream>>>(w, wh, wl, (int)(wn / 4));
        taylor_gemm_presplit_kernel<<<grid, 256, SMEM_BYTES, stream>>>(xh, xl, wh, wl, d0, out);
    } else {
        taylor_gemm_fused_kernel<<<grid, 256, SMEM_BYTES, stream>>>(x, w, d0, out);
    }
}
